// MultiHeadAttention_87909390614757
// MI455X (gfx1250) — matrix operand order repaired and checked by reading, not compiled, not run on hardware
//
#include <hip/hip_runtime.h>
#include <hip/hip_bf16.h>

// ---------------------------------------------------------------------------
// MI455X (gfx1250) fused MHA forward: bf16 WMMA (16x16x32) with f32 accum.
// wave32; WMMA operands here use contiguous K order (k = 16*(lane/16) + s) for A and B.
//   A-fragment (16x32 bf16): per lane two contiguous 16B runs:
//     lanes 0-15 : K = 0..7  and 8..15   (row m = lane&15)
//     lanes 16-31: K = 16..23 and 24..31
//   B-fragment (32x16 bf16): lane n=lane&15 holds col n; lanes 0-15 K=0..15,
//     lanes 16-31 K=16..31  -> 32 contiguous bytes per lane from a K-major row.
// GEMM K-loops are software-pipelined (prefetch next K-step while WMMAs run).
// ---------------------------------------------------------------------------

typedef __attribute__((ext_vector_type(16))) __bf16 v16bf;
typedef __attribute__((ext_vector_type(8)))  float  v8f;

#define HEADS   16
#define DMODEL  1024
#define DK      64
#define SEQ     2048
#define BATCH   4
#define ROWS    (BATCH * SEQ)          // 8192

__device__ __forceinline__ v8f wmma_bf16(v16bf a, v16bf b, v8f c) {
  return __builtin_amdgcn_wmma_f32_16x16x32_bf16(false, a, false, b, (short)0, c,
                                                 false, false);
}

// A-fragment (16x32, bf16): two b128 loads per lane.
__device__ __forceinline__ v16bf load_afrag_bf16(const __bf16* base, int ld,
                                                 int row0, int k0, int lane) {
  int m  = lane & 15;
  int kb = (lane & 16) ? 16 : 0;
  const __bf16* p = base + (long)(row0 + m) * ld + k0 + kb;
  union { v16bf v; uint4 q[2]; } r;
  r.q[0] = *(const uint4*)(p);
  r.q[1] = *(const uint4*)(p + 8);
  return r.v;
}

// A-fragment from fp32 row-major (convert on the fly): 4x float4 + pack.
__device__ __forceinline__ v16bf load_afrag_f32(const float* base, int ld,
                                                int row0, int k0, int lane) {
  int m  = lane & 15;
  int kb = (lane & 16) ? 16 : 0;
  const float* p = base + (long)(row0 + m) * ld + k0 + kb;
  float4 f0 = *(const float4*)(p);
  float4 f1 = *(const float4*)(p + 4);
  float4 f2 = *(const float4*)(p + 8);
  float4 f3 = *(const float4*)(p + 12);
  v16bf v;
  v[0]  = (__bf16)f0.x; v[1]  = (__bf16)f0.y; v[2]  = (__bf16)f0.z; v[3]  = (__bf16)f0.w;
  v[4]  = (__bf16)f1.x; v[5]  = (__bf16)f1.y; v[6]  = (__bf16)f1.z; v[7]  = (__bf16)f1.w;
  v[8]  = (__bf16)f2.x; v[9]  = (__bf16)f2.y; v[10] = (__bf16)f2.z; v[11] = (__bf16)f2.w;
  v[12] = (__bf16)f3.x; v[13] = (__bf16)f3.y; v[14] = (__bf16)f3.z; v[15] = (__bf16)f3.w;
  return v;
}

// ---------------------------------------------------------------------------
// Kernel 1: convert + transpose the four 1024x1024 fp32 weights to bf16.
// Wt[w][n][k] = W[w][k][n]
// ---------------------------------------------------------------------------
__global__ void cvt_weights(const float* __restrict__ Wq,
                            const float* __restrict__ Wk,
                            const float* __restrict__ Wv,
                            const float* __restrict__ Wo,
                            __bf16* __restrict__ Wt) {
  long idx = (long)blockIdx.x * 256 + threadIdx.x;     // 4 * 1M elements
  int  w   = (int)(idx >> 20);
  long e   = idx & ((1L << 20) - 1);
  int  k   = (int)(e >> 10);
  int  n   = (int)(e & 1023);
  const float* W = (w == 0) ? Wq : (w == 1) ? Wk : (w == 2) ? Wv : Wo;
  Wt[((long)w << 20) + (long)n * 1024 + k] = (__bf16)W[(long)k * 1024 + n];
}

// ---------------------------------------------------------------------------
// Kernel 2: projection GEMM  X(fp32 8192x1024) @ Wt(bf16, pre-transposed)
//           + bias, + RoPE (mode 0: Q/K -> [b][h][s][64])
//                    (mode 1: V      -> [b][h][64][s]  transposed)
// One wave computes a 32x64 strip (2 A-frags x 4 B-tiles = 8 WMMA / K-step),
// software-pipelined: next K-step operands are in flight during the WMMAs.
// ---------------------------------------------------------------------------
__global__ void proj_kernel(const float* __restrict__ X,
                            const __bf16* __restrict__ Wt,
                            const float* __restrict__ bias,
                            __bf16* __restrict__ Out, int mode) {
  int wid  = (int)(blockIdx.x * (blockDim.x >> 5)) + (threadIdx.x >> 5);
  int lane = threadIdx.x & 31;
  const int NB = DMODEL / 64;          // 16 column-strips
  int tm = wid / NB, tn = wid % NB;
  int row0 = tm * 32;

  v8f acc[2][4];
#pragma unroll
  for (int mi = 0; mi < 2; ++mi)
#pragma unroll
    for (int i = 0; i < 4; ++i)
#pragma unroll
      for (int c = 0; c < 8; ++c) acc[mi][i][c] = 0.0f;

  int n15 = lane & 15;
  int kbB = (lane & 16) ? 16 : 0;

  // prologue: load K-step 0
  v16bf a0 = load_afrag_f32(X, DMODEL, row0,      0, lane);
  v16bf a1 = load_afrag_f32(X, DMODEL, row0 + 16, 0, lane);
  v16bf b[4];
#pragma unroll
  for (int nt = 0; nt < 4; ++nt)
    b[nt] = *(const v16bf*)(Wt + (long)(tn * 64 + nt * 16 + n15) * DMODEL + kbB);

  for (int k0 = 0; k0 < DMODEL; k0 += 32) {
    int k1 = (k0 + 32) & (DMODEL - 1);      // wraps to 0 on last iter (harmless)
    v16bf na0 = load_afrag_f32(X, DMODEL, row0,      k1, lane);
    v16bf na1 = load_afrag_f32(X, DMODEL, row0 + 16, k1, lane);
    v16bf nb[4];
#pragma unroll
    for (int nt = 0; nt < 4; ++nt)
      nb[nt] = *(const v16bf*)(Wt + (long)(tn * 64 + nt * 16 + n15) * DMODEL + k1 + kbB);

#pragma unroll
    for (int nt = 0; nt < 4; ++nt) {
      acc[0][nt] = wmma_bf16(a0, b[nt], acc[0][nt]);
      acc[1][nt] = wmma_bf16(a1, b[nt], acc[1][nt]);
    }
    a0 = na0; a1 = na1;
#pragma unroll
    for (int nt = 0; nt < 4; ++nt) b[nt] = nb[nt];
  }

  int rbase = (lane & 16) ? 8 : 0;
#pragma unroll
  for (int mi = 0; mi < 2; ++mi) {
#pragma unroll
    for (int nt = 0; nt < 4; ++nt) {
      int col = tn * 64 + nt * 16 + n15;
      int h = col >> 6, d = col & 63;
      float bv = bias[col];
      float theta = __powf(10000.0f, -(float)(d & ~1) * (1.0f / 64.0f));
#pragma unroll
      for (int r = 0; r < 8; ++r) {
        int row = row0 + mi * 16 + rbase + r;
        int s = row & (SEQ - 1);
        int bb = row >> 11;
        float x = acc[mi][nt][r] + bv;
        if (mode == 0) {
          // rotary pair partner sits in the adjacent lane (col ^ 1)
          float xp = __shfl_xor(x, 1, 32);
          float sn, c;
          __sincosf((float)s * theta, &sn, &c);
          x = (d & 1) ? (x * c + xp * sn) : (x * c - xp * sn);
          long idx = (((long)bb * HEADS + h) * SEQ + s) * DK + d;
          Out[idx] = (__bf16)x;
        } else {
          long idx = (((long)bb * HEADS + h) * DK + d) * SEQ + s;   // V transposed
          Out[idx] = (__bf16)x;
        }
      }
    }
  }
}

// ---------------------------------------------------------------------------
// Kernel 3: flash attention per (b, h, 64-query block).
// 4 waves x 16 query rows; 32-key blocks; online softmax with 16-lane
// shuffle reductions; P re-layout C->A through per-wave LDS.
// Next block's K-fragments and this block's V-fragments are loaded before the
// softmax VALU section so their latency is hidden.
// ---------------------------------------------------------------------------
__global__ void attn_kernel(const __bf16* __restrict__ Qh,
                            const __bf16* __restrict__ Kh,
                            const __bf16* __restrict__ Vt,
                            const int* __restrict__ mask,
                            __bf16* __restrict__ Cc) {
  __shared__ __bf16 Pst[4][16][32];

  int wave = threadIdx.x >> 5;
  int lane = threadIdx.x & 31;
  int blk  = blockIdx.x;               // (b*16 + h)*32 + qb
  int qb   = blk & 31;
  int bh   = blk >> 5;
  int b    = bh >> 4;
  int h    = bh & 15;

  const __bf16* Qp = Qh + (long)bh * SEQ * DK;
  const __bf16* Kp = Kh + (long)bh * SEQ * DK;
  const __bf16* Vp = Vt + (long)bh * DK * SEQ;
  const int* mrow  = mask + b * SEQ;

  int q0 = qb * 64 + wave * 16;
  v16bf aq0 = load_afrag_bf16(Qp, DK, q0, 0, lane);
  v16bf aq1 = load_afrag_bf16(Qp, DK, q0, 32, lane);

  v8f accO[4], m8, l8;
#pragma unroll
  for (int i = 0; i < 4; ++i)
#pragma unroll
    for (int c = 0; c < 8; ++c) accO[i][c] = 0.0f;
#pragma unroll
  for (int c = 0; c < 8; ++c) { m8[c] = -3.0e38f; l8[c] = 0.0f; }

  int n15   = lane & 15;
  int kb16  = (lane & 16) ? 16 : 0;
  int kbA   = (lane & 16) ? 16 : 0;
  int rbase = (lane & 16) ? 8 : 0;

  // prologue: K-fragments for key block 0   bk[st][kt]
  v16bf bk[2][2];
#pragma unroll
  for (int st = 0; st < 2; ++st) {
    const __bf16* kp = Kp + (long)(st * 16 + n15) * DK + kb16;
    bk[st][0] = *(const v16bf*)(kp);
    bk[st][1] = *(const v16bf*)(kp + 32);
  }

  for (int key0 = 0; key0 < SEQ; key0 += 32) {
    // scores for the current 32-key block
    v8f S[2];
#pragma unroll
    for (int st = 0; st < 2; ++st) {
      v8f z;
#pragma unroll
      for (int c = 0; c < 8; ++c) z[c] = 0.0f;
      v8f s = wmma_bf16(aq1, bk[st][1], wmma_bf16(aq0, bk[st][0], z));
#pragma unroll
      for (int c = 0; c < 8; ++c) s[c] *= 0.125f;     // 1/sqrt(64)
      if (mrow[key0 + st * 16 + n15] == 0) {
#pragma unroll
        for (int c = 0; c < 8; ++c) s[c] = -1.0e9f;
      }
      S[st] = s;
    }

    // prefetch next block's K-fragments (wraps on last iter; harmless)
    int key1 = (key0 + 32) & (SEQ - 1);
#pragma unroll
    for (int st = 0; st < 2; ++st) {
      const __bf16* kp = Kp + (long)(key1 + st * 16 + n15) * DK + kb16;
      bk[st][0] = *(const v16bf*)(kp);
      bk[st][1] = *(const v16bf*)(kp + 32);
    }
    // V-fragments for the current block (consumed after softmax)
    v16bf bv[4];
#pragma unroll
    for (int nt = 0; nt < 4; ++nt)
      bv[nt] = *(const v16bf*)(Vp + (long)(nt * 16 + n15) * SEQ + key0 + kb16);

    // per-row max across the 16-lane key group
    v8f t;
#pragma unroll
    for (int c = 0; c < 8; ++c) t[c] = fmaxf(S[0][c], S[1][c]);
#pragma unroll
    for (int off = 1; off < 16; off <<= 1)
#pragma unroll
      for (int c = 0; c < 8; ++c) t[c] = fmaxf(t[c], __shfl_xor(t[c], off, 32));

    v8f mnew, scale;
#pragma unroll
    for (int c = 0; c < 8; ++c) {
      mnew[c]  = fmaxf(m8[c], t[c]);
      scale[c] = __expf(m8[c] - mnew[c]);
    }
#pragma unroll
    for (int st = 0; st < 2; ++st)
#pragma unroll
      for (int c = 0; c < 8; ++c) S[st][c] = __expf(S[st][c] - mnew[c]);

    v8f rs;
#pragma unroll
    for (int c = 0; c < 8; ++c) rs[c] = S[0][c] + S[1][c];
#pragma unroll
    for (int off = 1; off < 16; off <<= 1)
#pragma unroll
      for (int c = 0; c < 8; ++c) rs[c] += __shfl_xor(rs[c], off, 32);

#pragma unroll
    for (int c = 0; c < 8; ++c) {
      l8[c] = l8[c] * scale[c] + rs[c];
      m8[c] = mnew[c];
    }
#pragma unroll
    for (int nt = 0; nt < 4; ++nt)
#pragma unroll
      for (int c = 0; c < 8; ++c) accO[nt][c] *= scale[c];

    // stage P (C-layout) into per-wave LDS tile [16 rows][32 keys]
#pragma unroll
    for (int st = 0; st < 2; ++st)
#pragma unroll
      for (int r = 0; r < 8; ++r)
        Pst[wave][rbase + r][st * 16 + n15] = (__bf16)S[st][r];

    asm volatile("s_wait_dscnt 0" ::: "memory");   // wave-local RAW fence

    // reload P in A-fragment layout: two ds_load_b128 per lane
    union { v16bf v; uint4 q[2]; } ap;
    const __bf16* lp = &Pst[wave][n15][kbA];
    ap.q[0] = *(const uint4*)(lp);
    ap.q[1] = *(const uint4*)(lp + 8);
    // (same-wave LDS ops complete in order -> next iteration's stores cannot
    //  bypass these loads; no second fence needed)

#pragma unroll
    for (int nt = 0; nt < 4; ++nt)
      accO[nt] = wmma_bf16(ap.v, bv[nt], accO[nt]);
  }

  // normalize + store to concat [b][s][h*64+d] (bf16 A operand for out-proj)
#pragma unroll
  for (int nt = 0; nt < 4; ++nt) {
    int col = h * 64 + nt * 16 + n15;
#pragma unroll
    for (int r = 0; r < 8; ++r) {
      int row = q0 + rbase + r;
      float o = accO[nt][r] / l8[r];
      Cc[((long)b * SEQ + row) * DMODEL + col] = (__bf16)o;
    }
  }
}

// ---------------------------------------------------------------------------
// Kernel 4: output projection  concat(bf16) @ Wot(bf16) + bo -> fp32 out
// One wave computes a 32x64 strip; software-pipelined like proj_kernel.
// ---------------------------------------------------------------------------
__global__ void out_kernel(const __bf16* __restrict__ Cc,
                           const __bf16* __restrict__ Wot,
                           const float* __restrict__ bo,
                           float* __restrict__ Out) {
  int wid  = (int)(blockIdx.x * (blockDim.x >> 5)) + (threadIdx.x >> 5);
  int lane = threadIdx.x & 31;
  const int NB = DMODEL / 64;
  int tm = wid / NB, tn = wid % NB;
  int row0 = tm * 32;

  v8f acc[2][4];
#pragma unroll
  for (int mi = 0; mi < 2; ++mi)
#pragma unroll
    for (int i = 0; i < 4; ++i)
#pragma unroll
      for (int c = 0; c < 8; ++c) acc[mi][i][c] = 0.0f;

  int n15 = lane & 15;
  int kbB = (lane & 16) ? 16 : 0;

  v16bf a0 = load_afrag_bf16(Cc, DMODEL, row0,      0, lane);
  v16bf a1 = load_afrag_bf16(Cc, DMODEL, row0 + 16, 0, lane);
  v16bf b[4];
#pragma unroll
  for (int nt = 0; nt < 4; ++nt)
    b[nt] = *(const v16bf*)(Wot + (long)(tn * 64 + nt * 16 + n15) * DMODEL + kbB);

  for (int k0 = 0; k0 < DMODEL; k0 += 32) {
    int k1 = (k0 + 32) & (DMODEL - 1);
    v16bf na0 = load_afrag_bf16(Cc, DMODEL, row0,      k1, lane);
    v16bf na1 = load_afrag_bf16(Cc, DMODEL, row0 + 16, k1, lane);
    v16bf nb[4];
#pragma unroll
    for (int nt = 0; nt < 4; ++nt)
      nb[nt] = *(const v16bf*)(Wot + (long)(tn * 64 + nt * 16 + n15) * DMODEL + k1 + kbB);

#pragma unroll
    for (int nt = 0; nt < 4; ++nt) {
      acc[0][nt] = wmma_bf16(a0, b[nt], acc[0][nt]);
      acc[1][nt] = wmma_bf16(a1, b[nt], acc[1][nt]);
    }
    a0 = na0; a1 = na1;
#pragma unroll
    for (int nt = 0; nt < 4; ++nt) b[nt] = nb[nt];
  }

  int rbase = (lane & 16) ? 8 : 0;
#pragma unroll
  for (int mi = 0; mi < 2; ++mi) {
#pragma unroll
    for (int nt = 0; nt < 4; ++nt) {
      int col = tn * 64 + nt * 16 + n15;
      float bv = bo[col];
#pragma unroll
      for (int r = 0; r < 8; ++r) {
        int row = row0 + mi * 16 + rbase + r;
        Out[(long)row * DMODEL + col] = acc[mi][nt][r] + bv;
      }
    }
  }
}

// ---------------------------------------------------------------------------
extern "C" void kernel_launch(void* const* d_in, const int* in_sizes, int n_in,
                              void* d_out, int out_size, void* d_ws, size_t ws_size,
                              hipStream_t stream) {
  const float* q    = (const float*)d_in[0];
  const float* k    = (const float*)d_in[1];
  const float* v    = (const float*)d_in[2];
  const int*   mask = (const int*)  d_in[3];
  const float* Wq   = (const float*)d_in[4];
  const float* bq   = (const float*)d_in[5];
  const float* Wk   = (const float*)d_in[6];
  const float* bk   = (const float*)d_in[7];
  const float* Wv   = (const float*)d_in[8];
  const float* bv   = (const float*)d_in[9];
  const float* Wo   = (const float*)d_in[10];
  const float* bo   = (const float*)d_in[11];

  // workspace layout (bf16 elements)
  __bf16* Wt = (__bf16*)d_ws;                       // 4 x 1M  (8 MB)
  const long WSZ  = 1L << 20;                       // one weight matrix
  const long HSZ  = (long)BATCH * HEADS * SEQ * DK; // 8,388,608 elems (16 MB)
  __bf16* Qh = Wt + 4 * WSZ;
  __bf16* Kh = Qh + HSZ;
  __bf16* Vt = Kh + HSZ;
  __bf16* Cc = Vt + HSZ;

  // 1. weights -> bf16, transposed
  cvt_weights<<<(4 * 1024 * 1024) / 256, 256, 0, stream>>>(Wq, Wk, Wv, Wo, Wt);

  // 2. projections (+bias, +RoPE for Q/K; V stored transposed)
  const int projBlocks = (ROWS / 32) * (DMODEL / 64) / 8;   // 512
  proj_kernel<<<projBlocks, 256, 0, stream>>>(q, Wt + 0 * WSZ, bq, Qh, 0);
  proj_kernel<<<projBlocks, 256, 0, stream>>>(k, Wt + 1 * WSZ, bk, Kh, 0);
  proj_kernel<<<projBlocks, 256, 0, stream>>>(v, Wt + 2 * WSZ, bv, Vt, 1);

  // 3. flash attention: (b*h) * (2048/64) blocks of 4 waves
  attn_kernel<<<BATCH * HEADS * (SEQ / 64), 128, 0, stream>>>(Qh, Kh, Vt, mask, Cc);

  // 4. output projection -> fp32
  out_kernel<<<projBlocks, 256, 0, stream>>>(Cc, Wt + 3 * WSZ, bo, (float*)d_out);
}
